// EsmSelfAttentionWithContacts_8693013807630
// MI455X (gfx1250) — matrix operand order repaired and checked by reading, not compiled, not run on hardware
//
#include <hip/hip_runtime.h>
#include <hip/hip_bf16.h>

// ---------------------------------------------------------------------------
// ESM self-attention for MI455X (gfx1250), wave32 + WMMA bf16 (f32 accum).
//   Kernel R: one-shot rotary cos/sin table precompute (S x 32 angles).
//   Kernel 0: one-shot f32 -> bf16 conversion of X and stacked W.
//   Kernel 1: QKV projection from bf16 operands (pure b128 loads + WMMA in
//             the K-loop), bias + scale + rotary (table lookup) epilogue;
//             Q,K row-major bf16, V transposed bf16.
//   Kernel 2: flash attention per 16-query-row wave tile, fp32 softmax,
//             P staged through LDS (D-layout -> A-layout) for the PV WMMA.
// ---------------------------------------------------------------------------

typedef __attribute__((ext_vector_type(16))) __bf16         v16bf;
typedef __attribute__((ext_vector_type(16))) unsigned short v16us;
typedef __attribute__((ext_vector_type(8)))  float          v8f;

constexpr int Bn  = 4;
constexpr int Sn  = 1024;
constexpr int Hn  = 1280;
constexpr int NHn = 20;
constexpr int HDn = 64;

// log2(10000)/32 : inv_freq[i] = 2^(-i * K_ROT)
#define K_ROT 0.41524101186092027f

__device__ __forceinline__ unsigned short f2bf(float f) {
  unsigned int u = __float_as_uint(f);
  unsigned int r = u + 0x7FFFu + ((u >> 16) & 1u);   // round-to-nearest-even
  return (unsigned short)(r >> 16);
}

union BF16x16 { v16bf v; v16us u; uint4 q[2]; };

__device__ __forceinline__ v8f wmma_bf16(v16bf a, v16bf b, v8f c) {
  // (neg_a, A, neg_b, B, c_mod, C, reuse_a, reuse_b)
  return __builtin_amdgcn_wmma_f32_16x16x32_bf16(false, a, false, b,
                                                 (short)0, c, false, false);
}

// A-fragment: two contiguous 8-element (16B) bf16 chunks.
__device__ __forceinline__ v16bf ldbf_split(const unsigned short* p0,
                                            const unsigned short* p1) {
  BF16x16 r;
  r.q[0] = *(const uint4*)p0;
  r.q[1] = *(const uint4*)p1;
  return r.v;
}

// B-fragment: one contiguous 16-element (32B) bf16 chunk.
__device__ __forceinline__ v16bf ldbf32(const unsigned short* p) {
  BF16x16 r;
  r.q[0] = ((const uint4*)p)[0];
  r.q[1] = ((const uint4*)p)[1];
  return r.v;
}

// ---------------------------------------------------------------------------
// Kernel R: rotary table. ctab/stab: [S][32] fp32, angle = s * 10000^(-i/32).
// ---------------------------------------------------------------------------
__global__ __launch_bounds__(256) void rope_tab_kernel(float* __restrict__ ctab,
                                                       float* __restrict__ stab) {
  const int idx = blockIdx.x * blockDim.x + threadIdx.x;   // s*32 + i
  if (idx >= Sn * 32) return;
  const int i = idx & 31;
  const int s = idx >> 5;
  const float inv = exp2f(-(float)i * K_ROT);
  float sv, cv;
  sincosf((float)s * inv, &sv, &cv);
  ctab[idx] = cv;
  stab[idx] = sv;
}

// ---------------------------------------------------------------------------
// Kernel 0: f32 -> bf16 bulk conversion, 8 elements / thread.
// ---------------------------------------------------------------------------
__global__ __launch_bounds__(256) void cvt_bf16_kernel(
    const float* __restrict__ src, unsigned short* __restrict__ dst, int n8) {
  const int i = blockIdx.x * blockDim.x + threadIdx.x;
  if (i >= n8) return;
  const float4* p = (const float4*)(src + (size_t)i * 8);
  float4 a = p[0], b = p[1];
  union { uint4 q; unsigned short u[8]; } r;
  r.u[0] = f2bf(a.x); r.u[1] = f2bf(a.y); r.u[2] = f2bf(a.z); r.u[3] = f2bf(a.w);
  r.u[4] = f2bf(b.x); r.u[5] = f2bf(b.y); r.u[6] = f2bf(b.z); r.u[7] = f2bf(b.w);
  *(uint4*)(dst + (size_t)i * 8) = r.q;
}

// ---------------------------------------------------------------------------
// Kernel 1: QKV projection + bias + (scale, rotary) epilogue, bf16 operands.
// grid = (NH, M/128, 3), block = 128 (4 waves). Wave: 32 rows x 64 cols
// (8 accumulators, 8 WMMAs per 32-wide K step). which: 0=Q, 1=K, 2=V.
// ---------------------------------------------------------------------------
__global__ __launch_bounds__(128) void esm_qkv_rope_kernel(
    const unsigned short* __restrict__ Xb,   // [M][Hn] bf16
    const unsigned short* __restrict__ Wb,   // [3][Hn][Hn] bf16 (row = out ch)
    const float* __restrict__ bq, const float* __restrict__ bk,
    const float* __restrict__ bv,
    const float* __restrict__ ctab,     // [S][32]
    const float* __restrict__ stab,     // [S][32]
    unsigned short* __restrict__ qws,   // [B*NH][S][HD] bf16
    unsigned short* __restrict__ kws,   // [B*NH][S][HD] bf16
    unsigned short* __restrict__ vtws)  // [B*NH][HD][S] bf16 (transposed)
{
  const int h      = blockIdx.x;          // head
  const int mblk   = blockIdx.y;
  const int which  = blockIdx.z;
  const int wave   = threadIdx.x >> 5;
  const int lane   = threadIdx.x & 31;
  const int lane16 = lane & 15;
  const int kh     = lane >> 4;
  const int Mbase  = mblk * 128 + wave * 32;

  const unsigned short* W = Wb + (size_t)which * Hn * Hn;
  const float* bias = (which == 0) ? bq : (which == 1) ? bk : bv;

  // A fragment sources: rows Mbase+lane16, Mbase+16+lane16; chunks kh*16, kh*16+8.
  const unsigned short* x0 = Xb + (size_t)(Mbase +      lane16) * Hn + kh * 16;
  const unsigned short* x1 = Xb + (size_t)(Mbase + 16 + lane16) * Hn + kh * 16;
  // B fragment sources: W row n = h*64 + j*16 + lane16, K chunk at kh*16.
  const unsigned short* w0 = W + (size_t)(h * 64 +  0 + lane16) * Hn + kh * 16;
  const unsigned short* w1 = W + (size_t)(h * 64 + 16 + lane16) * Hn + kh * 16;
  const unsigned short* w2 = W + (size_t)(h * 64 + 32 + lane16) * Hn + kh * 16;
  const unsigned short* w3 = W + (size_t)(h * 64 + 48 + lane16) * Hn + kh * 16;

  v8f acc[2][4];
#pragma unroll
  for (int mt = 0; mt < 2; ++mt)
#pragma unroll
    for (int j = 0; j < 4; ++j) acc[mt][j] = (v8f){};

  for (int kb = 0; kb < Hn; kb += 32) {
    const v16bf A0 = ldbf_split(x0 + kb, x0 + kb + 8);
    const v16bf A1 = ldbf_split(x1 + kb, x1 + kb + 8);
    const v16bf B0 = ldbf32(w0 + kb);
    const v16bf B1 = ldbf32(w1 + kb);
    const v16bf B2 = ldbf32(w2 + kb);
    const v16bf B3 = ldbf32(w3 + kb);
    acc[0][0] = wmma_bf16(A0, B0, acc[0][0]);
    acc[1][0] = wmma_bf16(A1, B0, acc[1][0]);
    acc[0][1] = wmma_bf16(A0, B1, acc[0][1]);
    acc[1][1] = wmma_bf16(A1, B1, acc[1][1]);
    acc[0][2] = wmma_bf16(A0, B2, acc[0][2]);
    acc[1][2] = wmma_bf16(A1, B2, acc[1][2]);
    acc[0][3] = wmma_bf16(A0, B3, acc[0][3]);
    acc[1][3] = wmma_bf16(A1, B3, acc[1][3]);
  }

  const float bias0 = bias[h * 64 +  0 + lane16];
  const float bias1 = bias[h * 64 + 16 + lane16];
  const float bias2 = bias[h * 64 + 32 + lane16];
  const float bias3 = bias[h * 64 + 48 + lane16];

  if (which == 2) {
    // V: bias only, store transposed [bh][d][s].
#pragma unroll
    for (int mt = 0; mt < 2; ++mt) {
#pragma unroll
      for (int r = 0; r < 8; ++r) {
        const int m  = Mbase + mt * 16 + r + 8 * kh;
        const int bb = m >> 10;            // S = 1024
        const int s  = m & (Sn - 1);
        const size_t base = ((size_t)(bb * NHn + h) * HDn) * Sn + s;
        vtws[base + (size_t)( 0 + lane16) * Sn] = f2bf(acc[mt][0][r] + bias0);
        vtws[base + (size_t)(16 + lane16) * Sn] = f2bf(acc[mt][1][r] + bias1);
        vtws[base + (size_t)(32 + lane16) * Sn] = f2bf(acc[mt][2][r] + bias2);
        vtws[base + (size_t)(48 + lane16) * Sn] = f2bf(acc[mt][3][r] + bias3);
      }
    }
  } else {
    unsigned short* ows = (which == 0) ? qws : kws;
    const float scale = (which == 0) ? 0.125f : 1.0f;   // HD^-0.5 = 1/8
#pragma unroll
    for (int mt = 0; mt < 2; ++mt) {
#pragma unroll
      for (int r = 0; r < 8; ++r) {
        const int m  = Mbase + mt * 16 + r + 8 * kh;
        const int bb = m >> 10;
        const int s  = m & (Sn - 1);
        // rotary table lookups: i = lane16 (cols j=0,2) and 16+lane16 (j=1,3)
        const float ca = ctab[s * 32 +      lane16];
        const float sa = stab[s * 32 +      lane16];
        const float cb = ctab[s * 32 + 16 + lane16];
        const float sb = stab[s * 32 + 16 + lane16];
        const float e0 = (acc[mt][0][r] + bias0) * scale;
        const float e1 = (acc[mt][1][r] + bias1) * scale;
        const float e2 = (acc[mt][2][r] + bias2) * scale;
        const float e3 = (acc[mt][3][r] + bias3) * scale;
        // out[d] = x[d]*cos + rotate_half(x)[d]*sin, pairs d <-> d+-32
        const float y0 = e0 * ca - e2 * sa;
        const float y2 = e2 * ca + e0 * sa;
        const float y1 = e1 * cb - e3 * sb;
        const float y3 = e3 * cb + e1 * sb;
        const size_t base = ((size_t)(bb * NHn + h) * Sn + s) * HDn;
        ows[base +  0 + lane16] = f2bf(y0);
        ows[base + 16 + lane16] = f2bf(y1);
        ows[base + 32 + lane16] = f2bf(y2);
        ows[base + 48 + lane16] = f2bf(y3);
      }
    }
  }
}

// ---------------------------------------------------------------------------
// Kernel 2: flash attention. grid = (B*NH, S/64), block = 128 (4 waves),
// each wave owns a 16-query tile and streams keys in blocks of 32.
// ---------------------------------------------------------------------------
__global__ __launch_bounds__(128) void esm_flash_attn_kernel(
    const unsigned short* __restrict__ qws,
    const unsigned short* __restrict__ kws,
    const unsigned short* __restrict__ vtws,
    const float* __restrict__ mask,     // [B][1][1][S]
    float* __restrict__ out)            // [B][S][NH*HD]
{
  __shared__ __align__(32) unsigned short plds[4][16 * 32];  // per-wave P tile

  const int bh     = blockIdx.x;
  const int b      = bh / NHn;
  const int h      = bh - b * NHn;
  const int wave   = threadIdx.x >> 5;
  const int lane   = threadIdx.x & 31;
  const int lane16 = lane & 15;
  const int kh     = lane >> 4;
  const int qbase  = blockIdx.y * 64 + wave * 16;

  // Q A-fragments (d = 0..31 and 32..63), loaded once.
  const unsigned short* qrow = qws + ((size_t)bh * Sn + qbase + lane16) * HDn;
  const v16bf qa0 = ldbf_split(qrow +      kh * 16, qrow +  8 + kh * 16);
  const v16bf qa1 = ldbf_split(qrow + 32 + kh * 16, qrow + 40 + kh * 16);

  const unsigned short* kbh = kws  + (size_t)bh * Sn * HDn;
  const unsigned short* vbh = vtws + (size_t)bh * HDn * Sn;
  const float*          mrow = mask + (size_t)b * Sn;

  v8f o0 = {}, o1 = {}, o2 = {}, o3 = {};
  float mi[8], li[8];
#pragma unroll
  for (int r = 0; r < 8; ++r) { mi[r] = -3.0e38f; li[r] = 0.0f; }

  for (int kb = 0; kb < Sn; kb += 32) {
    // ---- scores: two 16x16 tiles over 32 keys --------------------------
    v8f s0 = {}, s1 = {};
    {
      const unsigned short* k0 = kbh + (size_t)(kb +      lane16) * HDn + kh * 16;
      const unsigned short* k1 = kbh + (size_t)(kb + 16 + lane16) * HDn + kh * 16;
      s0 = wmma_bf16(qa0, ldbf32(k0),      s0);
      s0 = wmma_bf16(qa1, ldbf32(k0 + 32), s0);
      s1 = wmma_bf16(qa0, ldbf32(k1),      s1);
      s1 = wmma_bf16(qa1, ldbf32(k1 + 32), s1);
    }
    const float mv0 = mrow[kb + lane16];
    const float mv1 = mrow[kb + 16 + lane16];

    // ---- online softmax (fp32, half-wave row reductions) ---------------
#pragma unroll
    for (int r = 0; r < 8; ++r) {
      const float sv0 = s0[r] + mv0;
      const float sv1 = s1[r] + mv1;
      float rowm = fmaxf(sv0, sv1);
      rowm = fmaxf(rowm, __shfl_xor(rowm, 1, 16));
      rowm = fmaxf(rowm, __shfl_xor(rowm, 2, 16));
      rowm = fmaxf(rowm, __shfl_xor(rowm, 4, 16));
      rowm = fmaxf(rowm, __shfl_xor(rowm, 8, 16));
      const float mnew  = fmaxf(mi[r], rowm);
      const float alpha = __expf(mi[r] - mnew);
      const float p0    = __expf(sv0 - mnew);
      const float p1    = __expf(sv1 - mnew);
      float rs = p0 + p1;
      rs += __shfl_xor(rs, 1, 16);
      rs += __shfl_xor(rs, 2, 16);
      rs += __shfl_xor(rs, 4, 16);
      rs += __shfl_xor(rs, 8, 16);
      li[r] = li[r] * alpha + rs;
      mi[r] = mnew;
      o0[r] *= alpha; o1[r] *= alpha; o2[r] *= alpha; o3[r] *= alpha;
      // stage P (D layout -> row-major 16x32 bf16 in LDS)
      plds[wave][(r + 8 * kh) * 32 +      lane16] = f2bf(p0);
      plds[wave][(r + 8 * kh) * 32 + 16 + lane16] = f2bf(p1);
    }
    // order per-wave DS writes before DS reads (CDNA5 split counters)
    asm volatile("s_wait_dscnt 0x0" ::: "memory");

    // ---- P @ V: read P as A-fragment, V^T columns are contiguous -------
    const unsigned short* pr = &plds[wave][lane16 * 32 + kh * 16];
    const v16bf pa = ldbf_split(pr, pr + 8);
    const unsigned short* v0 = vbh + (size_t)( 0 + lane16) * Sn + kb + kh * 16;
    const unsigned short* v1 = vbh + (size_t)(16 + lane16) * Sn + kb + kh * 16;
    const unsigned short* v2 = vbh + (size_t)(32 + lane16) * Sn + kb + kh * 16;
    const unsigned short* v3 = vbh + (size_t)(48 + lane16) * Sn + kb + kh * 16;
    o0 = wmma_bf16(pa, ldbf32(v0), o0);
    o1 = wmma_bf16(pa, ldbf32(v1), o1);
    o2 = wmma_bf16(pa, ldbf32(v2), o2);
    o3 = wmma_bf16(pa, ldbf32(v3), o3);
  }

  // ---- normalize and write [B][S][NH*HD] fp32 --------------------------
#pragma unroll
  for (int r = 0; r < 8; ++r) {
    const float inv = 1.0f / li[r];
    const int srow  = qbase + r + 8 * kh;
    float* orow = out + ((size_t)b * Sn + srow) * (NHn * HDn) + h * HDn;
    orow[ 0 + lane16] = o0[r] * inv;
    orow[16 + lane16] = o1[r] * inv;
    orow[32 + lane16] = o2[r] * inv;
    orow[48 + lane16] = o3[r] * inv;
  }
}

// ---------------------------------------------------------------------------
extern "C" void kernel_launch(void* const* d_in, const int* in_sizes, int n_in,
                              void* d_out, int out_size, void* d_ws, size_t ws_size,
                              hipStream_t stream) {
  const float* hidden = (const float*)d_in[0];
  const float* amask  = (const float*)d_in[1];
  const float* Wq     = (const float*)d_in[2];
  const float* bq     = (const float*)d_in[3];
  const float* Wk     = (const float*)d_in[4];
  const float* bk     = (const float*)d_in[5];
  const float* Wv     = (const float*)d_in[6];
  const float* bv     = (const float*)d_in[7];
  float* outp = (float*)d_out;

  const size_t qkv_elems = (size_t)Bn * NHn * Sn * HDn;   // 5,242,880
  const size_t x_elems   = (size_t)Bn * Sn * Hn;          // 5,242,880
  const size_t w_elems   = (size_t)Hn * Hn;               // 1,638,400
  const size_t tab_elems = (size_t)Sn * 32;               // 32,768

  unsigned short* qws  = (unsigned short*)d_ws;
  unsigned short* kws  = qws + qkv_elems;
  unsigned short* vtws = kws + qkv_elems;
  unsigned short* xbf  = vtws + qkv_elems;
  unsigned short* wbf  = xbf + x_elems;                   // [3][Hn][Hn]
  float*          ctab = (float*)(wbf + 3 * w_elems);     // 4B-aligned (even count)
  float*          stab = ctab + tab_elems;

  // ---- pass R/0: one-shot table + f32 -> bf16 conversions ---------------
  rope_tab_kernel<<<(int)((tab_elems + 255) / 256), 256, 0, stream>>>(ctab, stab);
  {
    const int nx8 = (int)(x_elems / 8);
    cvt_bf16_kernel<<<(nx8 + 255) / 256, 256, 0, stream>>>(hidden, xbf, nx8);
    const int nw8 = (int)(w_elems / 8);
    cvt_bf16_kernel<<<(nw8 + 255) / 256, 256, 0, stream>>>(Wq, wbf + 0 * w_elems, nw8);
    cvt_bf16_kernel<<<(nw8 + 255) / 256, 256, 0, stream>>>(Wk, wbf + 1 * w_elems, nw8);
    cvt_bf16_kernel<<<(nw8 + 255) / 256, 256, 0, stream>>>(Wv, wbf + 2 * w_elems, nw8);
  }

  // ---- pass 1: QKV + rotary --------------------------------------------
  dim3 g1(NHn, (Bn * Sn) / 128, 3);   // (20, 32, 3)
  esm_qkv_rope_kernel<<<g1, 128, 0, stream>>>(xbf, wbf, bq, bk, bv,
                                              ctab, stab, qws, kws, vtws);

  // ---- pass 2: flash attention -----------------------------------------
  dim3 g2(Bn * NHn, Sn / 64);         // (80, 16)
  esm_flash_attn_kernel<<<g2, 128, 0, stream>>>(qws, kws, vtws, amask, outp);
}
